// MAF_66529043415307
// MI455X (gfx1250) — matrix operand order repaired and checked by reading, not compiled, not run on hardware
//
#include <hip/hip_runtime.h>
#include <hip/hip_bf16.h>

// ---------------------------------------------------------------------------
// Inverse MAF for MI455X (gfx1250, wave32, WMMA).
//
// made(x) layers run as f16 WMMA (f32 accumulate); masked weights are
// pre-packed once per launch into the WMMA B-fragment lane layout so each
// lane's B fragment is one contiguous 32B load. Activations ping-pong in LDS.
// 128 sequential step kernels implement the autoregressive inversion.
// Final (s,t) columns are computed as a 2x512 dot spread over all 8 waves.
// ---------------------------------------------------------------------------

typedef __attribute__((ext_vector_type(16))) _Float16 v16h;
typedef __attribute__((ext_vector_type(8)))  float    v8f;

#define BATCH 2048
#define DIMX  128
#define HDIM  512

union HalfVec  { v16h v; uint4 q[2]; };
union Half8    { uint4 q; _Float16 h[8]; };

// ---------------------------------------------------------------------------
// One masked-MLP layer: 16-row batch tile (LDS, f16) x packed weights (global,
// f16) -> 16x512 output tile (LDS, f16) with bias + ReLU.
// Each of the 8 waves owns 4 contiguous 16-wide N-tiles.
// KB = K/32 (4 for layer 0, 16 for layers 1-2).
// ---------------------------------------------------------------------------
template<int KB>
__device__ __forceinline__ void made_layer(
    const _Float16 (*__restrict__ hin)[HDIM],
    _Float16 (*__restrict__ hout)[HDIM],
    const _Float16* __restrict__ Wp,
    const float* __restrict__ bias,
    int wave, int lane)
{
    const int m = lane & 15;   // A row handled by this lane
    const int g = lane >> 4;   // lane half (selects K sub-ranges)

    // Build all A fragments for this K extent (hoisted: reused by 4 N-tiles).
    // A k order used here (contiguous, as packed B): lane m    holds K = kb*32 + 0..15
    //                        lane m+16 holds K = kb*32 + 16..31
    v16h a[KB];
#pragma unroll
    for (int kb = 0; kb < KB; ++kb) {
        HalfVec u;
        u.q[0] = *(const uint4*)&hin[m][32 * kb + 16 * g];
        u.q[1] = *(const uint4*)&hin[m][32 * kb + 16 * g + 8];
        a[kb] = u.v;
    }

#pragma unroll
    for (int t = 0; t < 4; ++t) {
        const int nt = wave * 4 + t;
        v8f acc = {};
#pragma unroll
        for (int kb = 0; kb < KB; ++kb) {
            // Packed B fragment: 16 contiguous halves per lane (32B load).
            v16h b = *(const v16h*)(Wp + (size_t)((nt * KB + kb) * 32 + lane) * 16);
            acc = __builtin_amdgcn_wmma_f32_16x16x32_f16(
                false, a[kb], false, b, (short)0, acc, false, false);
        }
        // f32 C/D layout: vgpr r, lane l -> (m = 8*(l>>4)+r, n = l&15)
        const int n  = nt * 16 + m;
        const float bz = bias[n];
#pragma unroll
        for (int r = 0; r < 8; ++r) {
            float v = acc[r] + bz;
            v = v > 0.0f ? v : 0.0f;          // ReLU
            hout[8 * g + r][n] = (_Float16)v;
        }
    }
}

// ---------------------------------------------------------------------------
// One autoregressive step: full MADE forward for a 16-row tile, then update
// column `step` of x and accumulate log_det.
// ---------------------------------------------------------------------------
__global__ __launch_bounds__(256) void maf_step(
    const float* __restrict__ z,
    const float* __restrict__ b0, const float* __restrict__ b1,
    const float* __restrict__ b2, const float* __restrict__ b3,
    const _Float16* __restrict__ W0p, const _Float16* __restrict__ W1p,
    const _Float16* __restrict__ W2p, const float* __restrict__ W3t,
    float* __restrict__ x, float* __restrict__ logdet, int step)
{
    __shared__ __align__(16) _Float16 shA[16][HDIM];   // 16 KB
    __shared__ __align__(16) _Float16 shB[16][HDIM];   // 16 KB
    __shared__ float shP[8][16][2];                    // per-wave dot partials

    const int tid  = threadIdx.x;
    const int wave = tid >> 5;
    const int lane = tid & 31;
    const int rowbase = blockIdx.x * 16;

    // Stage x tile (16 x 128 f32 -> f16), vectorized: 2x float4 load + 1x 16B
    // LDS store per thread. Columns >= step are exactly zero by construction.
    {
        const int r  = tid >> 4;          // 0..15
        const int c0 = (tid & 15) * 8;    // 0..120
        const float* xp = &x[(size_t)(rowbase + r) * DIMX + c0];
        const float4 f0 = *(const float4*)xp;
        const float4 f1 = *(const float4*)(xp + 4);
        Half8 u;
        u.h[0] = (_Float16)f0.x; u.h[1] = (_Float16)f0.y;
        u.h[2] = (_Float16)f0.z; u.h[3] = (_Float16)f0.w;
        u.h[4] = (_Float16)f1.x; u.h[5] = (_Float16)f1.y;
        u.h[6] = (_Float16)f1.z; u.h[7] = (_Float16)f1.w;
        *(uint4*)&shA[r][c0] = u.q;
    }
    __syncthreads();

    made_layer<4>(shA, shB, W0p, b0, wave, lane);   // h0 = relu(x W0m + b0)
    __syncthreads();
    made_layer<16>(shB, shA, W1p, b1, wave, lane);  // h1
    __syncthreads();
    made_layer<16>(shA, shB, W2p, b2, wave, lane);  // h2
    __syncthreads();

    // Final layer: only output columns `step` (s) and `128+step` (t) needed.
    // Distribute the two 512-length dots over all 8 waves (64 K per wave,
    // 32 K per lane), vectorized loads, then an 8-way reduce via LDS.
    {
        const int r  = lane & 15;
        const int g  = lane >> 4;
        const int k0 = wave * 64 + g * 32;
        const float* w3s = W3t + (size_t)step * HDIM + k0;
        const float* w3t = W3t + (size_t)(DIMX + step) * HDIM + k0;
        float ss = 0.0f, tt = 0.0f;
#pragma unroll
        for (int c = 0; c < 4; ++c) {
            Half8 u;
            u.q = *(const uint4*)&shB[r][k0 + c * 8];
            const float4 ws0 = *(const float4*)(w3s + c * 8);
            const float4 ws1 = *(const float4*)(w3s + c * 8 + 4);
            const float4 wt0 = *(const float4*)(w3t + c * 8);
            const float4 wt1 = *(const float4*)(w3t + c * 8 + 4);
            float hv[8];
#pragma unroll
            for (int e = 0; e < 8; ++e) hv[e] = (float)u.h[e];
            ss += hv[0] * ws0.x + hv[1] * ws0.y + hv[2] * ws0.z + hv[3] * ws0.w
                + hv[4] * ws1.x + hv[5] * ws1.y + hv[6] * ws1.z + hv[7] * ws1.w;
            tt += hv[0] * wt0.x + hv[1] * wt0.y + hv[2] * wt0.z + hv[3] * wt0.w
                + hv[4] * wt1.x + hv[5] * wt1.y + hv[6] * wt1.z + hv[7] * wt1.w;
        }
        ss += __shfl_xor(ss, 16);
        tt += __shfl_xor(tt, 16);
        if (g == 0) { shP[wave][r][0] = ss; shP[wave][r][1] = tt; }
    }
    __syncthreads();

    if (wave == 0 && lane < 16) {
        const int r = lane;
        float ss = 0.0f, tt = 0.0f;
#pragma unroll
        for (int w = 0; w < 8; ++w) { ss += shP[w][r][0]; tt += shP[w][r][1]; }
        const int row = rowbase + r;
        const float s  = ss + b3[step];
        const float t  = tt + b3[DIMX + step];
        const float zi = z[(size_t)row * DIMX + step];
        x[(size_t)row * DIMX + step] = (zi - t) * expf(-s);
        logdet[row] -= s;
    }
}

// ---------------------------------------------------------------------------
// Prep kernels
// ---------------------------------------------------------------------------
__global__ void zero_out(float* __restrict__ p, int n)
{
    const int i = blockIdx.x * blockDim.x + threadIdx.x;
    if (i < n) p[i] = 0.0f;
}

// Pack W*mask (f32 row-major KxN, N=512) into WMMA B-fragment order, f16.
// B k order used here (contiguous, as A): lanes 0-15 hold K = kb*32 + h (h=half idx),
// lanes 16-31 hold K = kb*32 + 16 + h; column n = nt*16 + (lane&15).
// mask_mode 0: M0[k][n] = (k <= n%127);  1: M1[k][n] = (k%127 <= n%127).
__global__ void pack_w(const float* __restrict__ W, _Float16* __restrict__ Wp,
                       int Kd, int mask_mode)
{
    const int idx = blockIdx.x * blockDim.x + threadIdx.x;
    const int KB = Kd >> 5;
    const int total = 32 * KB * 512;
    if (idx >= total) return;
    const int h    = idx & 15;
    const int lane = (idx >> 4) & 31;
    const int frag = idx >> 9;
    const int nt = frag / KB, kb = frag - nt * KB;
    const int g = lane >> 4;
    const int k = 32 * kb + 16 * g + h;
    const int n = nt * 16 + (lane & 15);
    const float mval = (mask_mode == 0)
        ? ((k <= (n % 127)) ? 1.0f : 0.0f)
        : (((k % 127) <= (n % 127)) ? 1.0f : 0.0f);
    Wp[idx] = (_Float16)(W[(size_t)k * HDIM + n] * mval);
}

// W3t[j][k] = W3[k][j] * M3[k][j], f32, j in [0,256), k in [0,512).
// M3[k][j] = (k%127 < j%128).
__global__ void pack_w3(const float* __restrict__ W3, float* __restrict__ W3t)
{
    const int idx = blockIdx.x * blockDim.x + threadIdx.x;
    if (idx >= 256 * HDIM) return;
    const int k = idx & 511, j = idx >> 9;
    const float mval = ((k % 127) < (j % 128)) ? 1.0f : 0.0f;
    W3t[idx] = W3[(size_t)k * 256 + j] * mval;
}

// ---------------------------------------------------------------------------
extern "C" void kernel_launch(void* const* d_in, const int* in_sizes, int n_in,
                              void* d_out, int out_size, void* d_ws, size_t ws_size,
                              hipStream_t stream)
{
    const float* z  = (const float*)d_in[0];
    const float* W0 = (const float*)d_in[1];
    const float* b0 = (const float*)d_in[2];
    const float* W1 = (const float*)d_in[3];
    const float* b1 = (const float*)d_in[4];
    const float* W2 = (const float*)d_in[5];
    const float* b2 = (const float*)d_in[6];
    const float* W3 = (const float*)d_in[7];
    const float* b3 = (const float*)d_in[8];

    float* x      = (float*)d_out;                 // (2048, 128)
    float* logdet = x + (size_t)BATCH * DIMX;      // (2048,)

    // Workspace carve-out (~1.7 MB total).
    _Float16* w0p = (_Float16*)d_ws;               //  65536 halves
    _Float16* w1p = w0p + 65536;                   // 262144 halves
    _Float16* w2p = w1p + 262144;                  // 262144 halves
    float*    w3t = (float*)(w2p + 262144);        // 131072 floats

    zero_out<<<(BATCH * (DIMX + 1) + 255) / 256, 256, 0, stream>>>(
        x, BATCH * (DIMX + 1));
    pack_w<<<65536 / 256, 256, 0, stream>>>(W0, w0p, 128, 0);
    pack_w<<<262144 / 256, 256, 0, stream>>>(W1, w1p, 512, 1);
    pack_w<<<262144 / 256, 256, 0, stream>>>(W2, w2p, 512, 1);
    pack_w3<<<(256 * HDIM) / 256, 256, 0, stream>>>(W3, w3t);

    for (int i = 0; i < DIMX; ++i) {
        maf_step<<<BATCH / 16, 256, 0, stream>>>(
            z, b0, b1, b2, b3, w0p, w1p, w2p, w3t, x, logdet, i);
    }
}
